// ERM_23493471109771
// MI455X (gfx1250) — matrix operand order repaired and checked by reading, not compiled, not run on hardware
//
#include <hip/hip_runtime.h>
#include <cstdint>
#include <cstddef>

typedef __attribute__((ext_vector_type(16))) __bf16 v16bf;
typedef __attribute__((ext_vector_type(8)))  float  v8f;

#define NB   16
#define C1c  64
#define C2c  128
#define HSm  88
#define WSm  88
#define HLg  176
#define WLg  176
#define HW_S (HSm * WSm)   /* 7744  */
#define HW_L (HLg * WLg)   /* 30976 */

// Async global->LDS path (CDNA5): gate on builtin availability so the file
// always compiles; fallback is the synchronous VGPR->ds_store staging.
#if defined(__AMDGCN__) && \
    __has_builtin(__builtin_amdgcn_global_load_async_to_lds_b128) && \
    __has_builtin(__builtin_amdgcn_s_wait_asynccnt)
#define ASYNC_LDS 1
// builtin signature: (v4i addrspace(1)*, v4i addrspace(3)*, imm offset, imm cpol)
typedef int v4i_t __attribute__((__vector_size__(16)));
typedef __attribute__((address_space(1))) v4i_t g_v4i;
typedef __attribute__((address_space(3))) v4i_t l_v4i;
#else
#define ASYNC_LDS 0
#endif

template <int N>
__device__ __forceinline__ void wait_async() {
#if ASYNC_LDS
  __builtin_amdgcn_s_wait_asynccnt(N);
#endif
}

__device__ __forceinline__ unsigned short f2bf(float f) {
  unsigned int u = __float_as_uint(f);
  u += 0x7fffu + ((u >> 16) & 1u);
  return (unsigned short)(u >> 16);
}
__device__ __forceinline__ float bf2f(unsigned short h) {
  return __uint_as_float(((unsigned int)h) << 16);
}

// ---------------------------------------------------------------------------
// Implicit-GEMM convolution on v_wmma_f32_16x16x32_bf16.
//   Input: bf16 NHWC, halo-padded [B, H+KH-1, W+KW-1, CIN] (zero halo) so
//   every tap load is unconditional.
//   B tile (32K x 64N bf16, 4KB) double-buffered in LDS, filled with
//   GLOBAL_LOAD_ASYNC_TO_LDS_B128 (ASYNCcnt). One barrier per K-stage:
//     wait_asynccnt 0 -> barrier -> issue(t+1 into buf^1) -> WMMAs on buf.
//   Wave tiling: MT==4 -> 2x2 wave grid, each wave 32Mx32N (2 A-frags x
//   2 B-frags -> 4 WMMAs, halves LDS reads per WMMA). MT==1 -> 1 wave-column,
//   each wave one 16x16 tile.
//   Epilogue: y = acc*scale[c]+bias[c], optional ReLU; bf16 NHWC store into a
//   (possibly padded / wider-channel) buffer and/or f32 NCHW store.
// 4 waves/block.
// ---------------------------------------------------------------------------
template<int KH, int KW, int CIN, int MT>
__global__ __launch_bounds__(128)
void conv_wmma(const unsigned short* __restrict__ in, int H, int W,
               const unsigned short* __restrict__ wpack,
               const float* __restrict__ scale, const float* __restrict__ bias,
               int relu,
               unsigned short* __restrict__ out_bf, int outC, int coff,
               int outWp, int outPad, int outPix,
               float* __restrict__ out_f32, int coutReal)
{
  constexpr int KSTEPS = CIN / 32;
  constexpr int T      = KH * KW * KSTEPS; // pipeline stages
  constexpr int MW     = (MT >= 2) ? 2 : 1; // wave grid (M direction)
  constexpr int NW     = 4 / MW;            // wave grid (N direction)
  constexpr int MI     = MT / MW;           // 16-row m-tiles per wave
  constexpr int NI     = 4 / NW;            // 16-col n-subtiles per wave
  const int Wp = W + KW - 1;                // padded input width
  const int HW = H * W;
  __shared__ alignas(64) unsigned short ldsB[2 * 64 * 32];

  const int tid  = threadIdx.x;
  const int lane = tid & 31;
  const int wv   = tid >> 5;
  const int b    = blockIdx.z;
  const int tileBase = blockIdx.x * 64;

  // B-tile loader role: thread t -> (pixel nl, K-half)
  const int nl   = tid & 63;
  const int half = tid >> 6;
  const int p_l  = tileBase + nl;
  const int y_l  = p_l / W;
  const int x_l  = p_l - y_l * W;
  const size_t inImg = (size_t)b * (H + KH - 1) * Wp * CIN;

  const int m2 = wv % MW;
  const int n2 = wv / MW;

  v8f acc[MI * NI] = {};

  auto issue = [&](int t, int buf) {
    const int tap = t / KSTEPS;
    const int ks  = t - tap * KSTEPS;
    const int kh  = tap / KW;
    const int kw  = tap - kh * KW;
    const size_t goff = inImg +
        ((size_t)(y_l + kh) * Wp + (x_l + kw)) * CIN + ks * 32 + half * 16;
#if ASYNC_LDS
    g_v4i* gp = (g_v4i*)(unsigned long long)(uintptr_t)(in + goff);
    l_v4i* lp = (l_v4i*)(unsigned int)(uintptr_t)(
        &ldsB[buf * 2048 + nl * 32 + half * 16]);
    __builtin_amdgcn_global_load_async_to_lds_b128(gp, lp, 0, 0);
    __builtin_amdgcn_global_load_async_to_lds_b128(gp, lp, 16, 0);
#else
    const int4* s4 = reinterpret_cast<const int4*>(in + goff);
    int4 v0 = s4[0];
    int4 v1 = s4[1];
    int4* l4 = reinterpret_cast<int4*>(&ldsB[buf * 2048 + nl * 32 + half * 16]);
    l4[0] = v0;
    l4[1] = v1;
#endif
  };

  issue(0, 0);
  for (int t = 0; t < T; ++t) {
    const int buf = t & 1;
    const int tap = t / KSTEPS;
    const int ks  = t - tap * KSTEPS;

    // A fragments (global, packed lane order) -- overlaps the async wait
    v16bf afrag[MI];
#pragma unroll
    for (int mi = 0; mi < MI; ++mi) {
      const int mt = m2 * MI + mi;
      afrag[mi] = *reinterpret_cast<const v16bf*>(
          wpack + (((size_t)tap * KSTEPS + ks) * MT + mt) * 512 + lane * 16);
    }

    wait_async<0>();   // this wave's stage-t DMA complete
    __syncthreads();   // stage t published; all waves done reading buf^1 (t-1)
    if (t + 1 < T) issue(t + 1, buf ^ 1);  // overlaps stage-t WMMAs

    const int kb = (lane >> 4) * 8;
    v16bf bfrag[NI];
#pragma unroll
    for (int ni = 0; ni < NI; ++ni) {
      const int nloc = (n2 * NI + ni) * 16 + (lane & 15);
      uint4 bq[2];
      bq[0] = *reinterpret_cast<const uint4*>(&ldsB[buf * 2048 + nloc * 32 + kb]);
      bq[1] = *reinterpret_cast<const uint4*>(&ldsB[buf * 2048 + nloc * 32 + 16 + kb]);
      __builtin_memcpy(&bfrag[ni], bq, 32);
    }
#pragma unroll
    for (int mi = 0; mi < MI; ++mi) {
#pragma unroll
      for (int ni = 0; ni < NI; ++ni) {
        acc[mi * NI + ni] = __builtin_amdgcn_wmma_f32_16x16x32_bf16(
            false, afrag[mi], false, bfrag[ni], (short)0, acc[mi * NI + ni],
            false, false);
      }
    }
  }

  // Epilogue. C/D layout: lane -> col = lane&15, rows = (lane>>4)*8 + r
  const int col = lane & 15;
  const int rh  = lane >> 4;
#pragma unroll
  for (int mi = 0; mi < MI; ++mi) {
    const int cbase = (m2 * MI + mi) * 16 + rh * 8;
    float sc[8], bs[8];
#pragma unroll
    for (int r = 0; r < 8; ++r) { sc[r] = scale[cbase + r]; bs[r] = bias[cbase + r]; }
#pragma unroll
    for (int ni = 0; ni < NI; ++ni) {
      const int p = tileBase + (n2 * NI + ni) * 16 + col;
      const int y = p / W;
      const int x = p - y * W;
      float v[8];
#pragma unroll
      for (int r = 0; r < 8; ++r) {
        float t = acc[mi * NI + ni][r] * sc[r] + bs[r];
        if (relu) t = fmaxf(t, 0.0f);
        v[r] = t;
      }
      if (out_bf) {
        int4 st;
        st.x = (int)((unsigned)f2bf(v[0]) | ((unsigned)f2bf(v[1]) << 16));
        st.y = (int)((unsigned)f2bf(v[2]) | ((unsigned)f2bf(v[3]) << 16));
        st.z = (int)((unsigned)f2bf(v[4]) | ((unsigned)f2bf(v[5]) << 16));
        st.w = (int)((unsigned)f2bf(v[6]) | ((unsigned)f2bf(v[7]) << 16));
        const size_t opix =
            (size_t)b * outPix + (size_t)(y + outPad) * outWp + (x + outPad);
        *reinterpret_cast<int4*>(out_bf + opix * outC + coff + cbase) = st;
      }
      if (out_f32) {
#pragma unroll
        for (int r = 0; r < 8; ++r) {
          const int c = cbase + r;
          if (c < coutReal)
            out_f32[((size_t)b * coutReal + c) * HW + p] = v[r];
        }
      }
    }
  }
}

// ---------------------------------------------------------------------------
// Weight pre-pack into WMMA 16-bit A-fragment lane order.
// ISA layout (16-bit A 16x32): lanes 0-15 hold K in {0..7, 16..23},
// lanes 16-31 hold K in {8..15, 24..31}. Rows m >= cout are zero-padded.
// ---------------------------------------------------------------------------
__global__ void pack_w(const float* __restrict__ wsrc,
                       unsigned short* __restrict__ wp,
                       int cout, int cin, int khw, int mtT)
{
  const int idx = blockIdx.x * 256 + threadIdx.x;
  const int ksteps = cin / 32;
  const int total = khw * ksteps * mtT * 512;
  if (idx >= total) return;
  const int j    = idx & 15;
  const int lane = (idx >> 4) & 31;
  const int frag = idx >> 9;
  const int mt   = frag % mtT;
  const int rest = frag / mtT;
  const int ks   = rest % ksteps;
  const int tap  = rest / ksteps;
  const int m    = mt * 16 + (lane & 15);
  const int kh8  = (lane >> 4) * 8;
  const int k    = (j < 8) ? (kh8 + j) : (16 + kh8 + (j - 8));
  const int c    = ks * 32 + k;
  float v = 0.0f;
  if (m < cout) v = wsrc[((size_t)m * cin + c) * khw + tap];
  wp[idx] = f2bf(v);
}

// ---------------------------------------------------------------------------
// Elementwise / layout kernels
// ---------------------------------------------------------------------------

// f32 NCHW -> bf16 NHWC with zero halo of pad P
__global__ void to_nhwc_pad(const float* __restrict__ src,
                            unsigned short* __restrict__ dst,
                            int C, int H, int W, int P, long long total)
{
  long long idx = (long long)blockIdx.x * 256 + threadIdx.x;
  if (idx >= total) return;
  const int Wp = W + 2 * P, Hp = H + 2 * P;
  const int c = (int)(idx % C);
  long long rest = idx / C;
  const int xp = (int)(rest % Wp);
  rest /= Wp;
  const int yp = (int)(rest % Hp);
  const int b  = (int)(rest / Hp);
  const int y = yp - P, x = xp - P;
  float v = 0.0f;
  if (y >= 0 && y < H && x >= 0 && x < W)
    v = src[(((size_t)b * C + c) * H + y) * W + x];
  dst[idx] = f2bf(v);
}

// zero the halo of a padded bf16 NHWC buffer
__global__ void halo_zero(unsigned short* __restrict__ buf,
                          int H, int W, int P, int C, long long total)
{
  long long idx = (long long)blockIdx.x * 256 + threadIdx.x;
  if (idx >= total) return;
  const int Wp = W + 2 * P, Hp = H + 2 * P;
  long long rest = idx / C;
  const int xp = (int)(rest % Wp);
  rest /= Wp;
  const int yp = (int)(rest % Hp);
  if (yp >= P && yp < H + P && xp >= P && xp < W + P) return;  // interior
  buf[idx] = 0;
}

// bilinear 2x upsample (align_corners) of in_map, sigmoid of u and (1-u)
__global__ void maps_k(const float* __restrict__ im,
                       float* __restrict__ s, float* __restrict__ s1)
{
  int idx = blockIdx.x * 256 + threadIdx.x;
  if (idx >= NB * HW_L) return;
  int p = idx % HW_L;
  int b = idx / HW_L;
  int y = p / WLg, x = p % WLg;
  const float sf = 87.0f / 175.0f;
  float fy = y * sf, fx = x * sf;
  int y0 = (int)floorf(fy), x0 = (int)floorf(fx);
  int y1 = min(y0 + 1, HSm - 1), x1 = min(x0 + 1, WSm - 1);
  float wy = fy - y0, wx = fx - x0;
  const float* ip = im + (size_t)b * HW_S;
  float top = ip[y0 * WSm + x0] * (1.0f - wx) + ip[y0 * WSm + x1] * wx;
  float bot = ip[y1 * WSm + x0] * (1.0f - wx) + ip[y1 * WSm + x1] * wx;
  float u = top * (1.0f - wy) + bot * wy;
  s[idx]  = 1.0f / (1.0f + expf(-u));
  s1[idx] = 1.0f / (1.0f + expf(-(1.0f - u)));
}

// m = (dilate3x3(s) - s) + (dilate3x3(s1) - s1)
__global__ void incmap_k(const float* __restrict__ s,
                         const float* __restrict__ s1,
                         float* __restrict__ m)
{
  int idx = blockIdx.x * 256 + threadIdx.x;
  if (idx >= NB * HW_L) return;
  int p = idx % HW_L;
  int b = idx / HW_L;
  int y = p / WLg, x = p % WLg;
  const float* sp  = s  + (size_t)b * HW_L;
  const float* sp1 = s1 + (size_t)b * HW_L;
  float mx = -3.4e38f, mx1 = -3.4e38f;
  for (int dy = -1; dy <= 1; ++dy) {
    int yy = y + dy;
    if (yy < 0 || yy >= HLg) continue;
    for (int dx = -1; dx <= 1; ++dx) {
      int xx = x + dx;
      if (xx < 0 || xx >= WLg) continue;
      mx  = fmaxf(mx,  sp[yy * WLg + xx]);
      mx1 = fmaxf(mx1, sp1[yy * WLg + xx]);
    }
  }
  m[idx] = (mx - sp[p]) + (mx1 - sp1[p]);
}

// c_feature = cur_x * m -> bf16 NHWC padded [B, 178, 178, 64], zero halo
__global__ void cfeat_pad(const float* __restrict__ curx,
                          const float* __restrict__ m,
                          unsigned short* __restrict__ cf, long long total)
{
  long long idx = (long long)blockIdx.x * 256 + threadIdx.x;
  if (idx >= total) return;
  const int c = (int)(idx & 63);
  long long rest = idx >> 6;
  const int xp = (int)(rest % 178);
  rest /= 178;
  const int yp = (int)(rest % 178);
  const int b  = (int)(rest / 178);
  const int y = yp - 1, x = xp - 1;
  float v = 0.0f;
  if (y >= 0 && y < HLg && x >= 0 && x < WLg) {
    const int p = y * WLg + x;
    v = curx[((size_t)b * C1c + c) * HW_L + p] * m[(size_t)b * HW_L + p];
  }
  cf[idx] = f2bf(v);
}

// bilinear 2x upsample (align_corners) of dep (bf16 NHWC 64ch @88x88)
// into channels [0,64) of the padded 128ch concat buffer [B,178,178,128]
__global__ void upsample_dep(const unsigned short* __restrict__ ds,
                             unsigned short* __restrict__ cc)
{
  long long idx = (long long)blockIdx.x * 256 + threadIdx.x;
  if (idx >= (long long)NB * HW_L * C1c) return;
  const int c = (int)(idx & 63);
  long long rest = idx >> 6;
  const int p = (int)(rest % HW_L);
  const int b = (int)(rest / HW_L);
  const int y = p / WLg, x = p % WLg;
  const float sf = 87.0f / 175.0f;
  float fy = y * sf, fx = x * sf;
  int y0 = (int)floorf(fy), x0 = (int)floorf(fx);
  int y1 = min(y0 + 1, HSm - 1), x1 = min(x0 + 1, WSm - 1);
  float wy = fy - y0, wx = fx - x0;
  const unsigned short* dp = ds + (size_t)b * HW_S * C1c;
  float a00 = bf2f(dp[((size_t)y0 * WSm + x0) * C1c + c]);
  float a01 = bf2f(dp[((size_t)y0 * WSm + x1) * C1c + c]);
  float a10 = bf2f(dp[((size_t)y1 * WSm + x0) * C1c + c]);
  float a11 = bf2f(dp[((size_t)y1 * WSm + x1) * C1c + c]);
  float top = a00 * (1.0f - wx) + a01 * wx;
  float bot = a10 * (1.0f - wx) + a11 * wx;
  float v = top * (1.0f - wy) + bot * wy;
  cc[((size_t)b * 31684 + (size_t)(y + 1) * 178 + (x + 1)) * C2c + c] = f2bf(v);
}

// fold BN (+ conv bias) into per-channel scale/bias
__global__ void bn_affine_k(const float* g, const float* be, const float* m,
                            const float* v, const float* convb,
                            float* sc, float* bi, int n)
{
  int c = threadIdx.x;
  if (c < n) {
    float a = g[c] * rsqrtf(v[c] + 1e-5f);
    sc[c] = a;
    bi[c] = (convb[c] - m[c]) * a + be[c];
  }
}

__global__ void beta_affine_k(const float* beta, const float* convb,
                              float* sc, float* bi, int n)
{
  int c = threadIdx.x;
  if (c < n) { sc[c] = beta[0]; bi[c] = beta[0] * convb[c]; }
}

__global__ void out_affine_k(const float* outb, float* sc, float* bi)
{
  int c = threadIdx.x;
  if (c < 16) { sc[c] = 1.0f; bi[c] = (c == 0) ? outb[0] : 0.0f; }
}

// ---------------------------------------------------------------------------
extern "C" void kernel_launch(void* const* d_in, const int* in_sizes, int n_in,
                              void* d_out, int out_size, void* d_ws,
                              size_t ws_size, hipStream_t stream)
{
  (void)in_sizes; (void)n_in; (void)out_size; (void)ws_size;

  const float* cur_x   = (const float*)d_in[0];
  const float* dep_x   = (const float*)d_in[1];
  const float* in_map  = (const float*)d_in[2];
  const float* up_w    = (const float*)d_in[3];
  const float* up_b    = (const float*)d_in[4];
  const float* conv2_w = (const float*)d_in[5];
  const float* conv2_b = (const float*)d_in[6];
  const float* d1_w    = (const float*)d_in[7];
  const float* d1_b    = (const float*)d_in[8];
  const float* d2_w    = (const float*)d_in[9];
  const float* d2_b    = (const float*)d_in[10];
  const float* d3_w    = (const float*)d_in[11];
  const float* d3_b    = (const float*)d_in[12];
  const float* out_w   = (const float*)d_in[13];
  const float* out_b   = (const float*)d_in[14];
  const float* beta    = (const float*)d_in[15];
  const float* up_g  = (const float*)d_in[16];
  const float* up_be = (const float*)d_in[17];
  const float* up_m  = (const float*)d_in[18];
  const float* up_v  = (const float*)d_in[19];
  const float* d1_g  = (const float*)d_in[20];
  const float* d1_be = (const float*)d_in[21];
  const float* d1_m  = (const float*)d_in[22];
  const float* d1_v  = (const float*)d_in[23];
  const float* d2_g  = (const float*)d_in[24];
  const float* d2_be = (const float*)d_in[25];
  const float* d2_m  = (const float*)d_in[26];
  const float* d2_v  = (const float*)d_in[27];
  const float* d3_g  = (const float*)d_in[28];
  const float* d3_be = (const float*)d_in[29];
  const float* d3_m  = (const float*)d_in[30];
  const float* d3_v  = (const float*)d_in[31];

  char* ws = (char*)d_ws;
  size_t cur = 0;
  auto alloc = [&](size_t bytes) -> char* {
    char* p = ws + cur;
    cur = (cur + bytes + 255) & ~(size_t)255;
    return p;
  };

  // padded NHWC bf16 buffers (pad = consumer conv radius)
  unsigned short* depx_bf = (unsigned short*)alloc((size_t)NB * 94 * 94 * C2c * 2);   // 7x7
  unsigned short* dep_s   = (unsigned short*)alloc((size_t)NB * HW_S * C1c * 2);      // unpadded
  unsigned short* cc      = (unsigned short*)alloc((size_t)NB * 178 * 178 * C2c * 2); // 3x3
  unsigned short* cfeat   = (unsigned short*)alloc((size_t)NB * 178 * 178 * C1c * 2); // 3x3
  unsigned short* x1      = (unsigned short*)alloc((size_t)NB * 178 * 178 * C1c * 2); // 3x3
  unsigned short* x2      = cfeat;  // cfeat dead after conv2
  unsigned short* ref_bf  = cc;     // concat dead after d1; 182*182*64 fits
  float* smap  = (float*)alloc((size_t)NB * HW_L * 4);
  float* s1map = (float*)alloc((size_t)NB * HW_L * 4);
  float* mmap  = (float*)alloc((size_t)NB * HW_L * 4);

  unsigned short* wp_up  = (unsigned short*)alloc((size_t)49 * 4 * 4 * 512 * 2);
  unsigned short* wp_c2  = (unsigned short*)alloc((size_t)9 * 2 * 4 * 512 * 2);
  unsigned short* wp_d1  = (unsigned short*)alloc((size_t)9 * 4 * 4 * 512 * 2);
  unsigned short* wp_d2  = (unsigned short*)alloc((size_t)9 * 2 * 4 * 512 * 2);
  unsigned short* wp_d3  = (unsigned short*)alloc((size_t)9 * 2 * 4 * 512 * 2);
  unsigned short* wp_out = (unsigned short*)alloc((size_t)49 * 2 * 1 * 512 * 2);

  float* sc_up  = (float*)alloc(64 * 4);  float* bi_up  = (float*)alloc(64 * 4);
  float* sc_c2  = (float*)alloc(64 * 4);  float* bi_c2  = (float*)alloc(64 * 4);
  float* sc_d1  = (float*)alloc(64 * 4);  float* bi_d1  = (float*)alloc(64 * 4);
  float* sc_d2  = (float*)alloc(64 * 4);  float* bi_d2  = (float*)alloc(64 * 4);
  float* sc_d3  = (float*)alloc(64 * 4);  float* bi_d3  = (float*)alloc(64 * 4);
  float* sc_o   = (float*)alloc(64 * 4);  float* bi_o   = (float*)alloc(64 * 4);

  float* refine_out = (float*)d_out;                            // [16,64,176,176]
  float* outmap_out = (float*)d_out + (size_t)NB * C1c * HW_L;  // [16,1,176,176]

  auto cdiv = [](long long a, long long b) { return (unsigned)((a + b - 1) / b); };

  // weight packing + fused epilogue coefficients
  pack_w<<<cdiv(49LL * 4 * 4 * 512, 256), 256, 0, stream>>>(up_w,    wp_up,  64, 128, 49, 4);
  pack_w<<<cdiv( 9LL * 2 * 4 * 512, 256), 256, 0, stream>>>(conv2_w, wp_c2,  64,  64,  9, 4);
  pack_w<<<cdiv( 9LL * 4 * 4 * 512, 256), 256, 0, stream>>>(d1_w,    wp_d1,  64, 128,  9, 4);
  pack_w<<<cdiv( 9LL * 2 * 4 * 512, 256), 256, 0, stream>>>(d2_w,    wp_d2,  64,  64,  9, 4);
  pack_w<<<cdiv( 9LL * 2 * 4 * 512, 256), 256, 0, stream>>>(d3_w,    wp_d3,  64,  64,  9, 4);
  pack_w<<<cdiv(49LL * 2 * 1 * 512, 256), 256, 0, stream>>>(out_w,   wp_out,  1,  64, 49, 1);

  bn_affine_k<<<1, 64, 0, stream>>>(up_g, up_be, up_m, up_v, up_b, sc_up, bi_up, 64);
  bn_affine_k<<<1, 64, 0, stream>>>(d1_g, d1_be, d1_m, d1_v, d1_b, sc_d1, bi_d1, 64);
  bn_affine_k<<<1, 64, 0, stream>>>(d2_g, d2_be, d2_m, d2_v, d2_b, sc_d2, bi_d2, 64);
  bn_affine_k<<<1, 64, 0, stream>>>(d3_g, d3_be, d3_m, d3_v, d3_b, sc_d3, bi_d3, 64);
  beta_affine_k<<<1, 64, 0, stream>>>(beta, conv2_b, sc_c2, bi_c2, 64);
  out_affine_k<<<1, 16, 0, stream>>>(out_b, sc_o, bi_o);

  // elementwise path
  to_nhwc_pad<<<cdiv((long long)NB * 94 * 94 * C2c, 256), 256, 0, stream>>>(
      dep_x, depx_bf, C2c, HSm, WSm, 3, (long long)NB * 94 * 94 * C2c);
  maps_k<<<cdiv((long long)NB * HW_L, 256), 256, 0, stream>>>(in_map, smap, s1map);
  incmap_k<<<cdiv((long long)NB * HW_L, 256), 256, 0, stream>>>(smap, s1map, mmap);
  cfeat_pad<<<cdiv((long long)NB * 178 * 178 * C1c, 256), 256, 0, stream>>>(
      cur_x, mmap, cfeat, (long long)NB * 178 * 178 * C1c);

  // up conv (7x7, 128->64, 88x88, BN+ReLU) -> dep_s (unpadded)
  {
    dim3 g(HW_S / 64, 1, NB);
    conv_wmma<7, 7, 128, 4><<<g, 128, 0, stream>>>(
        depx_bf, HSm, WSm, wp_up, sc_up, bi_up, 1,
        dep_s, 64, 0, WSm, 0, HW_S, nullptr, 64);
  }
  // concat buffer: zero halo (all 128 ch), upsampled dep -> ch [0,64)
  halo_zero<<<cdiv((long long)NB * 178 * 178 * C2c, 256), 256, 0, stream>>>(
      cc, HLg, WLg, 1, C2c, (long long)NB * 178 * 178 * C2c);
  upsample_dep<<<cdiv((long long)NB * HW_L * C1c, 256), 256, 0, stream>>>(dep_s, cc);

  // conv2 (3x3, 64->64): beta*fn -> concat ch [64,128)
  {
    dim3 g(HW_L / 64, 1, NB);
    conv_wmma<3, 3, 64, 4><<<g, 128, 0, stream>>>(
        cfeat, HLg, WLg, wp_c2, sc_c2, bi_c2, 0,
        cc, 128, 64, 178, 1, 31684, nullptr, 64);
  }
  // d1 (3x3, 128->64, BN+ReLU) -> x1 (padded)
  {
    dim3 g(HW_L / 64, 1, NB);
    conv_wmma<3, 3, 128, 4><<<g, 128, 0, stream>>>(
        cc, HLg, WLg, wp_d1, sc_d1, bi_d1, 1,
        x1, 64, 0, 178, 1, 31684, nullptr, 64);
  }
  halo_zero<<<cdiv((long long)NB * 178 * 178 * C1c, 256), 256, 0, stream>>>(
      x1, HLg, WLg, 1, C1c, (long long)NB * 178 * 178 * C1c);
  // ref_bf halo (aliases cc; cc dead after d1)
  halo_zero<<<cdiv((long long)NB * 182 * 182 * C1c, 256), 256, 0, stream>>>(
      ref_bf, HLg, WLg, 3, C1c, (long long)NB * 182 * 182 * C1c);
  // d2 (3x3, 64->64, BN+ReLU) -> x2 (padded; halo zeroed by cfeat_pad)
  {
    dim3 g(HW_L / 64, 1, NB);
    conv_wmma<3, 3, 64, 4><<<g, 128, 0, stream>>>(
        x1, HLg, WLg, wp_d2, sc_d2, bi_d2, 1,
        x2, 64, 0, 178, 1, 31684, nullptr, 64);
  }
  // d3 (3x3, 64->64, BN+ReLU) -> refine2: f32 NCHW to d_out + bf16 padded
  {
    dim3 g(HW_L / 64, 1, NB);
    conv_wmma<3, 3, 64, 4><<<g, 128, 0, stream>>>(
        x2, HLg, WLg, wp_d3, sc_d3, bi_d3, 1,
        ref_bf, 64, 0, 182, 3, 33124, refine_out, 64);
  }
  // out conv (7x7, 64->1) -> output_map (f32 NCHW)
  {
    dim3 g(HW_L / 64, 1, NB);
    conv_wmma<7, 7, 64, 1><<<g, 128, 0, stream>>>(
        ref_bf, HLg, WLg, wp_out, sc_o, bi_o, 0,
        nullptr, 64, 0, 182, 3, 33124, outmap_out, 1);
  }
}
